// MIL_aggregation_14362370638472
// MI455X (gfx1250) — matrix operand order repaired and checked by reading, not compiled, not run on hardware
//
#include <hip/hip_runtime.h>
#include <hip/hip_fp16.h>

typedef __attribute__((ext_vector_type(16))) _Float16 v16h;
typedef __attribute__((ext_vector_type(8)))  _Float16 v8h;
typedef __attribute__((ext_vector_type(4)))  _Float16 v4h;
typedef __attribute__((ext_vector_type(8)))  float    v8f;
typedef __attribute__((ext_vector_type(4)))  float    v4f;

#define NBAGS 2048
#define NI    196      // instances per bag
#define NPAD  208      // 13 * 16
#define DD    768
#define HH    128
#define AA    64
#define KT1   24       // 768/32 k-steps GEMM1
#define KT2   4        // 128/32 k-steps GEMM2
#define NT1   16       // 256/16 n-tiles GEMM1 (w1|w2 concat)
#define NT2   4        // 64/16  n-tiles GEMM2
#define LNEPS 1e-5f

// LDS element strides (padded: 16B-aligned rows, conflict-free 16-lane access)
#define XSF 36         // xbuf  [3][64][XSF] f32  (144B row stride)
#define ES  136        // embbuf[208][ES]    f16  (272B row stride)
#define HS  264        // hbuf  [64][HS]     f32
#define TS  76         // tbuf  [208][TS]    f32
#define XBUFELT (64 * XSF)   // floats per x staging buffer (2304)

// dynamic LDS layout (bytes)
#define OFF_EMB   0
#define OFF_SBUF  (NPAD * ES * 2)                 // 56576
#define OFF_SCR   (OFF_SBUF + NPAD * 4)           // 57408
#define OFF_XBUF  OFF_SCR                         // 3 * 9216 = 27648 B
#define OFF_HBUF  (OFF_SCR + 3 * XBUFELT * 4)     // 85056
#define OFF_TBUF  OFF_SCR                         // 63232 B (phase 2 only)
#define SMEM_BYTES (OFF_HBUF + 64 * HS * 4)       // 152640

// ---------------------------------------------------------------------------
// Prep: repack w1|w2 and wa (f32 row-major [K][N]) into per-lane f16 WMMA
// B-fragment layout: [nt][kt][lane][16 halves] -> one contiguous 32B chunk
// per lane (two global_load_b128).
//   B 32x16 f16 mapping: n = ncol0 + lane%16 ; k = kt*32 + (lane>=16?8:0) + (h&7) + (h>=8?16:0)
// ---------------------------------------------------------------------------
__global__ void pack_weights_kernel(const float* __restrict__ w1,
                                    const float* __restrict__ w2,
                                    const float* __restrict__ wa,
                                    _Float16* __restrict__ wp1,
                                    _Float16* __restrict__ wpa) {
  int idx = blockIdx.x * 256 + threadIdx.x;
  const int tot1 = NT1 * KT1 * 32 * 16;           // 196608
  if (idx < tot1) {
    int h    = idx & 15;
    int lane = (idx >> 4) & 31;
    int kt   = (idx >> 9) % KT1;
    int nt   = idx / (KT1 * 512);
    int n = (nt & 7) * 16 + (lane & 15);
    int k = kt * 32 + ((lane >> 4) << 3) + (h & 7) + ((h >> 3) << 4);
    const float* src = (nt < 8) ? w1 : w2;
    wp1[idx] = (_Float16)src[k * HH + n];
  }
  if (idx < NT2 * KT2 * 32 * 16) {                // 8192
    int h    = idx & 15;
    int lane = (idx >> 4) & 31;
    int kt   = (idx >> 9) % KT2;
    int nt   = idx / (KT2 * 512);
    int n = nt * 16 + (lane & 15);
    int k = kt * 32 + ((lane >> 4) << 3) + (h & 7) + ((h >> 3) << 4);
    wpa[idx] = (_Float16)wa[k * AA + n];
  }
}

// Async copy of 32B (8 f32) global -> LDS, tracked by ASYNCcnt.
__device__ __forceinline__ void async_copy8(const float* gsrc, float* ldst,
                                            bool valid) {
  if (valid) {
    unsigned l0 = (unsigned)(size_t)ldst;  // flat LDS ptr low 32b = LDS offset
    asm volatile("global_load_async_to_lds_b128 %0, %1, off"
                 :: "v"(l0), "v"(gsrc) : "memory");
    unsigned l1 = l0 + 16;
    const float* g1 = gsrc + 4;
    asm volatile("global_load_async_to_lds_b128 %0, %1, off"
                 :: "v"(l1), "v"(g1) : "memory");
  }
}

// A-fragment (16x32, f16) built from an f32 LDS tile (stride XSF floats).
// ISA layout: lane L: m = L%16 ; k = b+{0..7}, b+16+{0..7}, b = (L>=16?8:0)
__device__ __forceinline__ v16h load_afrag_f32(const float* base, int mrow,
                                               int lane) {
  const float* p = base + (mrow + (lane & 15)) * XSF + ((lane >> 4) << 3);
  v4f f0 = *(const v4f*)p;
  v4f f1 = *(const v4f*)(p + 4);
  v4f f2 = *(const v4f*)(p + 16);
  v4f f3 = *(const v4f*)(p + 20);
  v16h a;
#pragma unroll
  for (int j = 0; j < 4; ++j) {
    a[j]      = (_Float16)f0[j];
    a[4 + j]  = (_Float16)f1[j];
    a[8 + j]  = (_Float16)f2[j];
    a[12 + j] = (_Float16)f3[j];
  }
  return a;
}

// A-fragment (16x32, f16) from an f16 LDS tile (stride `stride` halves).
__device__ __forceinline__ v16h load_afrag_f16(const _Float16* base, int mrow,
                                               int stride, int kcol, int lane) {
  const _Float16* p =
      base + (mrow + (lane & 15)) * stride + kcol + ((lane >> 4) << 3);
  v8h lo = *(const v8h*)p;
  v8h hi = *(const v8h*)(p + 16);
  return __builtin_shufflevector(lo, hi, 0, 1, 2, 3, 4, 5, 6, 7,
                                 8, 9, 10, 11, 12, 13, 14, 15);
}

__device__ __forceinline__ float wave_sum(float v) {
#pragma unroll
  for (int off = 16; off > 0; off >>= 1) v += __shfl_xor(v, off, 32);
  return v;
}

// ---------------------------------------------------------------------------
// Fused MIL kernel: one block (256 threads / 8 waves) per bag.
// ---------------------------------------------------------------------------
__global__ __launch_bounds__(256) void mil_fused_kernel(
    const float* __restrict__ x, const float* __restrict__ label,
    const float* __restrict__ b1, const float* __restrict__ g1,
    const float* __restrict__ be1, const float* __restrict__ b2,
    const float* __restrict__ g2, const float* __restrict__ be2,
    const float* __restrict__ ba, const float* __restrict__ ga,
    const float* __restrict__ bea, const float* __restrict__ wb,
    const float* __restrict__ bb, const _Float16* __restrict__ wp1,
    const _Float16* __restrict__ wpa, float* __restrict__ out) {
  extern __shared__ char smem[];
  _Float16* embbuf = (_Float16*)(smem + OFF_EMB);   // [208][ES] f16
  float*    sbuf   = (float*)(smem + OFF_SBUF);     // [208] scores/weights
  float*    xbuf   = (float*)(smem + OFF_XBUF);     // [3][64][XSF] f32
  float*    hbuf   = (float*)(smem + OFF_HBUF);     // [64][HS] f32
  float*    tbuf   = (float*)(smem + OFF_TBUF);     // [208][TS] f32

  const int tid  = threadIdx.x;
  const int lane = tid & 31;
  const int wid  = tid >> 5;
  const int bag  = blockIdx.x;
  const float* xb = x + (size_t)bag * NI * DD;

  // zero the padded rows of embbuf (rows 196..207)
  for (int i = tid; i < (NPAD - NI) * ES; i += 256)
    embbuf[NI * ES + i] = (_Float16)0.f;

  // per-lane LN params for the H=128 dims (lane covers cols lane*4..+3)
  v4f b1v  = *(const v4f*)(b1 + lane * 4);
  v4f g1v  = *(const v4f*)(g1 + lane * 4);
  v4f be1v = *(const v4f*)(be1 + lane * 4);
  v4f b2v  = *(const v4f*)(b2 + lane * 4);
  v4f g2v  = *(const v4f*)(g2 + lane * 4);
  v4f be2v = *(const v4f*)(be2 + lane * 4);

  const int wm = wid >> 2;  // 0..1 : which pair of M-tiles
  const int wn = wid & 3;   // 0..3 : which group of 4 N-tiles

  // this thread's share of the x staging tile: 8 f32 of one row per step
  const int sr  = tid >> 2;          // 0..63 staged row
  const int sc0 = (tid & 3) * 8;     // 0,8,16,24 staged col

  // ------------------- Phase 1: GEMM1 + LN + tanh*sigmoid -------------------
  for (int chunk = 0; chunk < 4; ++chunk) {
    const int row0  = chunk * 64;
    const bool vld  = (row0 + sr) < NI;
    const float* gb = xb + (size_t)(row0 + sr) * DD + sc0;
    float* lb       = xbuf + sr * XSF + sc0;

    // prologue: prefetch k-step 0 into buffer 0
    async_copy8(gb, lb, vld);

    v8f acc[2][4] = {};
    for (int kt = 0; kt < KT1; ++kt) {
      // prefetch next k-step (distance 1, triple buffer), then drain current
      if (kt + 1 < KT1) {
        async_copy8(gb + (kt + 1) * 32, lb + ((kt + 1) % 3) * XBUFELT, vld);
        asm volatile("s_wait_asynccnt 2" ::: "memory");
      } else {
        asm volatile("s_wait_asynccnt 0" ::: "memory");
      }
      __syncthreads();  // single barrier per k-step (triple buffering)

      const float* cb = xbuf + (kt % 3) * XBUFELT;
      v16h a0 = load_afrag_f32(cb, (wm * 2 + 0) * 16, lane);
      v16h a1 = load_afrag_f32(cb, (wm * 2 + 1) * 16, lane);
#pragma unroll
      for (int j = 0; j < 4; ++j) {
        int nt = wn * 4 + j;
        v16h bf = *(const v16h*)(wp1 + (((nt * KT1 + kt) * 32 + lane) << 4));
        acc[0][j] = __builtin_amdgcn_wmma_f32_16x16x32_f16(
            false, a0, false, bf, (short)0, acc[0][j], false, false);
        acc[1][j] = __builtin_amdgcn_wmma_f32_16x16x32_f16(
            false, a1, false, bf, (short)0, acc[1][j], false, false);
      }
    }
    __syncthreads();  // xbuf reads done before any reuse; acc ready to spill

    // spill h = x@[w1|w2] tiles to LDS (C layout: row = v + (lane>=16?8:0))
#pragma unroll
    for (int i = 0; i < 2; ++i) {
      int rbase = (wm * 2 + i) * 16 + ((lane >> 4) << 3);
#pragma unroll
      for (int j = 0; j < 4; ++j) {
        int col = (wn * 4 + j) * 16 + (lane & 15);
#pragma unroll
        for (int v = 0; v < 8; ++v)
          hbuf[(rbase + v) * HS + col] = acc[i][j][v];
      }
    }
    __syncthreads();

    // per-row LN (H=128) + tanh * sigmoid gate -> embbuf (f16)
    for (int rr = 0; rr < 8; ++rr) {
      int r  = wid * 8 + rr;
      int gr = row0 + r;
      if (gr >= NI) break;  // uniform per wave
      const float* hrow = hbuf + r * HS;
      v4f h1 = *(const v4f*)(hrow + lane * 4) + b1v;
      v4f h2 = *(const v4f*)(hrow + HH + lane * 4) + b2v;
      float mu1 = wave_sum(h1[0] + h1[1] + h1[2] + h1[3]) * (1.f / 128.f);
      float mu2 = wave_sum(h2[0] + h2[1] + h2[2] + h2[3]) * (1.f / 128.f);
      v4f d1 = h1 - mu1, d2 = h2 - mu2;
      float q1 = wave_sum(d1[0]*d1[0] + d1[1]*d1[1] + d1[2]*d1[2] + d1[3]*d1[3]);
      float q2 = wave_sum(d2[0]*d2[0] + d2[1]*d2[1] + d2[2]*d2[2] + d2[3]*d2[3]);
      float rs1 = rsqrtf(q1 * (1.f / 128.f) + LNEPS);
      float rs2 = rsqrtf(q2 * (1.f / 128.f) + LNEPS);
      v4f n1 = d1 * rs1 * g1v + be1v;
      v4f n2 = d2 * rs2 * g2v + be2v;
      v4h e;
#pragma unroll
      for (int j = 0; j < 4; ++j) {
        float t  = tanhf(n1[j]);
        float sg = 1.f / (1.f + __expf(-n2[j]));
        e[j] = (_Float16)(t * sg);
      }
      *(v4h*)(embbuf + gr * ES + lane * 4) = e;
    }
    __syncthreads();
  }

  // ------------------- Phase 2: GEMM2 (emb @ wa) + LN + score ---------------
  for (int mt = wid; mt < 13; mt += 8) {
    v8f acc2[4] = {};
    for (int kt = 0; kt < KT2; ++kt) {
      v16h a = load_afrag_f16(embbuf, mt * 16, ES, kt * 32, lane);
#pragma unroll
      for (int j = 0; j < 4; ++j) {
        v16h bf = *(const v16h*)(wpa + (((j * KT2 + kt) * 32 + lane) << 4));
        acc2[j] = __builtin_amdgcn_wmma_f32_16x16x32_f16(
            false, a, false, bf, (short)0, acc2[j], false, false);
      }
    }
    int rbase = mt * 16 + ((lane >> 4) << 3);
#pragma unroll
    for (int j = 0; j < 4; ++j)
#pragma unroll
      for (int v = 0; v < 8; ++v)
        tbuf[(rbase + v) * TS + j * 16 + (lane & 15)] = acc2[j][v];
  }
  __syncthreads();

  // per-row LN over A=64 + relu, then score s = relu(t' . wb + bb)
  {
    float bav0 = ba[lane * 2],  bav1 = ba[lane * 2 + 1];
    float gav0 = ga[lane * 2],  gav1 = ga[lane * 2 + 1];
    float bev0 = bea[lane * 2], bev1 = bea[lane * 2 + 1];
    float wbv0 = wb[lane * 2],  wbv1 = wb[lane * 2 + 1];
    float bbv  = bb[0];
    for (int r = wid; r < NI; r += 8) {
      float t0 = tbuf[r * TS + lane * 2] + bav0;
      float t1 = tbuf[r * TS + lane * 2 + 1] + bav1;
      float mu = wave_sum(t0 + t1) * (1.f / 64.f);
      float d0 = t0 - mu, d1 = t1 - mu;
      float rs = rsqrtf(wave_sum(d0 * d0 + d1 * d1) * (1.f / 64.f) + LNEPS);
      float n0 = fmaxf(d0 * rs * gav0 + bev0, 0.f);
      float n1 = fmaxf(d1 * rs * gav1 + bev1, 0.f);
      float sc = wave_sum(n0 * wbv0 + n1 * wbv1);
      if (lane == 0) sbuf[r] = fmaxf(sc + bbv, 0.f);
    }
  }
  __syncthreads();

  // ------------------- Phase 3: softmax * sigma(label) ----------------------
  if (wid == 0) {
    float vals[7];
    float mx = -1e30f;
#pragma unroll
    for (int i = 0; i < 7; ++i) {
      int n = i * 32 + lane;
      float v = (n < NI) ? sbuf[n] : -1e30f;
      vals[i] = v;
      mx = fmaxf(mx, v);
    }
#pragma unroll
    for (int off = 16; off > 0; off >>= 1) mx = fmaxf(mx, __shfl_xor(mx, off, 32));
    float sum = 0.f;
#pragma unroll
    for (int i = 0; i < 7; ++i) {
      int n = i * 32 + lane;
      float e = (n < NI) ? __expf(vals[i] - mx) : 0.f;
      vals[i] = e;
      sum += e;
    }
    sum = wave_sum(sum);
    float inv = 1.f / sum;
#pragma unroll
    for (int i = 0; i < 7; ++i) {
      int n = i * 32 + lane;
      if (n < NI) {
        float lb = label[(size_t)bag * NI + n];
        float theta = (lb > 0.5f) ? (1.f - lb) : lb;
        float sg = (1.f - theta) * (1.f - theta);
        sbuf[n] = sg * vals[i] * inv;
      }
    }
  }
  __syncthreads();

  // ------------------- Phase 4: out[b,d] = sum_n w[n] * emb[n,d] ------------
  if (tid < HH) {
    float o = 0.f;
    for (int n = 0; n < NI; ++n)
      o += sbuf[n] * (float)embbuf[n * ES + tid];
    out[(size_t)bag * HH + tid] = o;
  }
}

// ---------------------------------------------------------------------------
extern "C" void kernel_launch(void* const* d_in, const int* in_sizes, int n_in,
                              void* d_out, int out_size, void* d_ws,
                              size_t ws_size, hipStream_t stream) {
  const float* x     = (const float*)d_in[0];
  const float* label = (const float*)d_in[1];
  const float* w1    = (const float*)d_in[2];
  const float* b1    = (const float*)d_in[3];
  const float* g1    = (const float*)d_in[4];
  const float* be1   = (const float*)d_in[5];
  const float* w2    = (const float*)d_in[6];
  const float* b2    = (const float*)d_in[7];
  const float* g2    = (const float*)d_in[8];
  const float* be2   = (const float*)d_in[9];
  const float* wa    = (const float*)d_in[10];
  const float* ba    = (const float*)d_in[11];
  const float* ga    = (const float*)d_in[12];
  const float* bea   = (const float*)d_in[13];
  const float* wb    = (const float*)d_in[14];
  const float* bb    = (const float*)d_in[15];

  _Float16* wp1 = (_Float16*)d_ws;                       // 196608 halves
  _Float16* wpa = (_Float16*)((char*)d_ws + 393216);     // 8192 halves

  (void)hipFuncSetAttribute((const void*)mil_fused_kernel,
                            hipFuncAttributeMaxDynamicSharedMemorySize,
                            SMEM_BYTES);

  pack_weights_kernel<<<768, 256, 0, stream>>>(w1, w2, wa, wp1, wpa);
  mil_fused_kernel<<<NBAGS, 256, SMEM_BYTES, stream>>>(
      x, label, b1, g1, be1, b2, g2, be2, ba, ga, bea, wb, bb, wp1, wpa,
      (float*)d_out);
}
